// MSA_47253230190881
// MI455X (gfx1250) — matrix operand order repaired and checked by reading, not compiled, not run on hardware
//
#include <hip/hip_runtime.h>

// ---------------------------------------------------------------- constants
#define B_  4
#define T_  2048
#define H_  16
#define HD_ 64
#define E_  1024

#define USE_ASYNC 1   // gfx1250 global_load_async_to_lds_b128 staging

typedef __attribute__((ext_vector_type(16))) __bf16 v16bf;
typedef __attribute__((ext_vector_type(8)))  __bf16 v8bf;
typedef __attribute__((ext_vector_type(8)))  float  v8f;

union AFrag { v16bf v; v8bf h[2]; __bf16 e[16]; };

// A-matrix fragment (16x32 bf16) per ISA 7.12.2:
// lane half==0 holds K = base+[0..7] and base+[16..23]; half==1 holds +8.
static __device__ inline v16bf load_afrag(const __bf16* row, int base, int half) {
  AFrag u;
  u.h[0] = *(const v8bf*)(row + base + half * 8);
  u.h[1] = *(const v8bf*)(row + base + 16 + half * 8);
  return u.v;
}

// B-matrix fragment (32x16 bf16): lane = N column, K in table order, K stride ld.
static __device__ inline v16bf load_bfrag(const __bf16* p, int ld, int half) {
  AFrag u;
#pragma unroll
  for (int s = 0; s < 8; ++s) {
    u.e[s]     = p[(half * 8 + s) * ld];
    u.e[8 + s] = p[(16 + half * 8 + s) * ld];
  }
  return u.v;
}

static __device__ inline v8f wmma_bf16(v16bf a, v16bf b, v8f c) {
  return __builtin_amdgcn_wmma_f32_16x16x32_bf16(false, a, false, b, (short)0, c,
                                                 false, false);
}

// ---- async staging helpers: copy 32B of K and 32B of V (per thread) to LDS.
// Generic pointers to __shared__ carry the LDS byte offset in addr[31:0]
// (aperture layout, ISA 10.2), so truncation yields the VDST LDS address.
static __device__ inline void stage_tile(__bf16* kdst, __bf16* vdst,
                                         const __bf16* kg, const __bf16* vg) {
#if USE_ASYNC
  asm volatile(
      "global_load_async_to_lds_b128 %0, %2, off\n\t"
      "global_load_async_to_lds_b128 %0, %2, off offset:16\n\t"
      "global_load_async_to_lds_b128 %1, %3, off\n\t"
      "global_load_async_to_lds_b128 %1, %3, off offset:16"
      :: "v"((unsigned)(size_t)kdst), "v"((unsigned)(size_t)vdst),
         "v"((unsigned long long)(size_t)kg), "v"((unsigned long long)(size_t)vg)
      : "memory");
#else
  *(v8bf*)kdst       = *(const v8bf*)kg;
  *(v8bf*)(kdst + 8) = *(const v8bf*)(kg + 8);
  *(v8bf*)vdst       = *(const v8bf*)vg;
  *(v8bf*)(vdst + 8) = *(const v8bf*)(vg + 8);
#endif
}
static __device__ inline void wait_async_le4() {
#if USE_ASYNC
  asm volatile("s_wait_asynccnt 0x4" ::: "memory");
#endif
}
static __device__ inline void wait_async_0() {
#if USE_ASYNC
  asm volatile("s_wait_asynccnt 0x0" ::: "memory");
#endif
}

// ----------------------------------------- 0a) x relayout to per-head bf16
// xbf[((b*H+h)*T + t)*64 + d] = x[(b*T+t)*E + h*64 + d]
__global__ void msa_xbf_kernel(const float* __restrict__ x,
                               __bf16* __restrict__ xbf) {
  int idx = blockIdx.x * blockDim.x + threadIdx.x;   // [0, B*T*E)
  int d = idx & 63;
  int t = (idx >> 6) & (T_ - 1);
  int h = (idx >> 17) & 15;
  int b = idx >> 21;
  xbf[idx] = (__bf16)x[((size_t)(b * T_ + t)) * E_ + h * HD_ + d];
}

// ----------------------------------------- 0b) W_qkv bf16 convert, same [192][64] layout
__global__ void msa_wqt_kernel(const float* __restrict__ Wqkv,
                               __bf16* __restrict__ wqt) {
  int idx = blockIdx.x * blockDim.x + threadIdx.x;   // [0, 64*192)
  int e = idx % 192, d = idx / 192;
  wqt[(size_t)e * HD_ + d] = (__bf16)Wqkv[(size_t)e * HD_ + d];
}

// ------------------------------------------- 0c) W_proj bf16 convert, same [E][E] layout
__global__ void msa_wt_kernel(const float* __restrict__ Wp, __bf16* __restrict__ wt) {
  int idx = blockIdx.x * blockDim.x + threadIdx.x;   // [0, E*E)
  int f = idx / E_, e = idx % E_;
  wt[(size_t)f * E_ + e] = (__bf16)Wp[(size_t)f * E_ + e];
}

// ---------------------------------------------------- 1) QKV projection (WMMA)
// per (b,h): [T x 64] @ [64 x 192]; wave computes a 16x192 strip (24 WMMAs).
__launch_bounds__(128)
__global__ void msa_qkv_kernel(const __bf16* __restrict__ xbf,
                               const __bf16* __restrict__ wqt,
                               const float* __restrict__ bqkv,
                               __bf16* __restrict__ qb, __bf16* __restrict__ kb,
                               __bf16* __restrict__ vb) {
  const int tid  = threadIdx.x;
  const int w    = tid >> 5;
  const int lane = tid & 31;
  const int half = lane >> 4;
  const int lrow = lane & 15;
  const int blk  = blockIdx.x;                 // 64 bh * 32 t-tiles
  const int bh = blk >> 5;
  const int t0 = (blk & 31) * 64 + w * 16;

  const __bf16* arow = xbf + ((size_t)bh * T_ + t0 + lrow) * HD_;
  v16bf a0 = load_afrag(arow, 0, half);        // dims 0..31
  v16bf a1 = load_afrag(arow, 32, half);       // dims 32..63

#pragma unroll
  for (int j = 0; j < 12; ++j) {               // 12 n-tiles of 16 -> 192 cols
    v8f acc = {};
    acc = wmma_bf16(a0, load_afrag(wqt + (size_t)(j * 16 + lrow) * HD_, 0, half), acc);
    acc = wmma_bf16(a1, load_afrag(wqt + (size_t)(j * 16 + lrow) * HD_, 32, half), acc);
    const int e = j * 16 + lrow;
    const float bias = bqkv[e];
    __bf16* dst;
    int d;
    float scale = 1.0f;
    if (j < 4)      { dst = qb; d = e;       scale = 0.125f; }  // fold 1/sqrt(hd)
    else if (j < 8) { dst = kb; d = e - 64;  }
    else            { dst = vb; d = e - 128; }
#pragma unroll
    for (int r = 0; r < 8; ++r) {
      int t = t0 + r + 8 * half;
      dst[((size_t)bh * T_ + t) * HD_ + d] = (__bf16)((acc[r] + bias) * scale);
    }
  }
}

// --------------------------------------------------- 2) flash attention (WMMA)
// grid: B*H*(T/64) blocks; 128 threads = 4 waves; wave owns 16 query rows.
// Scores computed TRANSPOSED (S^T = K @ Q^T) so the S^T accumulator maps
// directly onto the P A-fragment layout: softmax is register-resident.
__launch_bounds__(128)
__global__ void msa_attn_kernel(const __bf16* __restrict__ qb,
                                const __bf16* __restrict__ kb,
                                const __bf16* __restrict__ vb,
                                __bf16* __restrict__ ob) {
  __shared__ __bf16 Kt[2][32][64];     // double-buffered K tile [key][dim]
  __shared__ __bf16 Vt[2][32][64];     // double-buffered V tile [key][dim]
  __shared__ __bf16 Qt[4][64][16];     // per-wave Q^T: [dim][query]

  const int tid  = threadIdx.x;
  const int w    = tid >> 5;
  const int lane = tid & 31;
  const int half = lane >> 4;
  const int lrow = lane & 15;

  const int blk = blockIdx.x;
  const int qt  = blk & 31;            // T/64 = 32 query strips
  const int bh  = blk >> 5;
  const int b = bh >> 4, h = bh & 15;
  const int q0 = qt * 64 + w * 16;

  const __bf16* kbh = kb + (size_t)bh * T_ * HD_;
  const __bf16* vbh = vb + (size_t)bh * T_ * HD_;
  const int key  = tid >> 2;
  const int dseg = (tid & 3) * 16;

  // prologue: kick off tile 0 into buffer 0
  stage_tile(&Kt[0][key][dseg], &Vt[0][key][dseg],
             kbh + (size_t)key * HD_ + dseg, vbh + (size_t)key * HD_ + dseg);

  // stage Q^T once (wave-local), then keep its B-fragments in registers
  {
    const __bf16* qrow = qb + ((size_t)bh * T_ + q0 + lrow) * HD_ + half * 32;
    v8bf qv[4];
#pragma unroll
    for (int c = 0; c < 4; ++c) qv[c] = *(const v8bf*)(qrow + c * 8);
#pragma unroll
    for (int d = 0; d < 32; ++d) Qt[w][half * 32 + d][lrow] = qv[d / 8][d % 8];
  }
  __syncthreads();
  v16bf qbf0 = load_afrag(qb + ((size_t)bh * T_ + q0 + lrow) * HD_, 0, half);   // B-frag of Q^T, dims 0..31
  v16bf qbf1 = load_afrag(qb + ((size_t)bh * T_ + q0 + lrow) * HD_, 32, half);  // dims 32..63

  v8f o0 = {}, o1 = {}, o2 = {}, o3 = {};          // O accumulator 16x64
  float m_prev = -1e30f, l_prev = 0.f;             // per-query state (dup x2)

  for (int kt = 0; kt < T_; kt += 32) {
    const int cur = (kt >> 5) & 1;
    __syncthreads();                      // prev compute done before overwrite
    if (kt + 32 < T_) {
      const int nb = cur ^ 1;
      const __bf16* kg = kbh + (size_t)(kt + 32 + key) * HD_ + dseg;
      const __bf16* vg = vbh + (size_t)(kt + 32 + key) * HD_ + dseg;
      stage_tile(&Kt[nb][key][dseg], &Vt[nb][key][dseg], kg, vg);
      wait_async_le4();                   // tile `cur` complete (this wave)
    } else {
      wait_async_0();
    }
    __syncthreads();                      // tile `cur` complete (all waves)

    // ---- preload ALL fragments for this tile into distinct registers so
    //      ds_load latency overlaps with WMMA/softmax instead of lockstep waits
    const __bf16* krow0 = &Kt[cur][lrow][0];
    const __bf16* krow1 = &Kt[cur][16 + lrow][0];
    v16bf ka00 = load_afrag(krow0, 0, half);
    v16bf ka01 = load_afrag(krow0, 32, half);
    v16bf ka10 = load_afrag(krow1, 0, half);
    v16bf ka11 = load_afrag(krow1, 32, half);
    v16bf vb0 = load_bfrag(&Vt[cur][0][lrow], 64, half);
    v16bf vb1 = load_bfrag(&Vt[cur][0][16 + lrow], 64, half);
    v16bf vb2 = load_bfrag(&Vt[cur][0][32 + lrow], 64, half);
    v16bf vb3 = load_bfrag(&Vt[cur][0][48 + lrow], 64, half);

    // ---- S^T(32x16) = K(32x64) @ Q^T(64x16), two 16-key sub-tiles
    v8f acc0 = {}, acc1 = {};
    acc0 = wmma_bf16(ka00, qbf0, acc0);
    acc0 = wmma_bf16(ka01, qbf1, acc0);
    acc1 = wmma_bf16(ka10, qbf0, acc1);
    acc1 = wmma_bf16(ka11, qbf1, acc1);

    // ---- register-resident online softmax (lane l <-> l^16 share a query)
    float m_new = m_prev;
#pragma unroll
    for (int j = 0; j < 8; ++j) m_new = fmaxf(m_new, fmaxf(acc0[j], acc1[j]));
    m_new = fmaxf(m_new, __shfl_xor(m_new, 16, 32));
    float alpha = __expf(m_prev - m_new);
    AFrag pu;                             // P in A-fragment layout, in place
    float ssum = 0.f;
#pragma unroll
    for (int j = 0; j < 8; ++j) {
      float p0 = __expf(acc0[j] - m_new);
      float p1 = __expf(acc1[j] - m_new);
      ssum += p0 + p1;
      pu.e[j]     = (__bf16)p0;           // keys half*8 + j
      pu.e[8 + j] = (__bf16)p1;           // keys 16 + half*8 + j
    }
    ssum += __shfl_xor(ssum, 16, 32);
    l_prev = l_prev * alpha + ssum;
    m_prev = m_new;

    // ---- rescale O rows (alpha for query r+8*half lives in lane r+8*half)
#pragma unroll
    for (int r = 0; r < 8; ++r) {
      float a = __shfl(alpha, r + 8 * half, 32);
      o0[r] *= a; o1[r] *= a; o2[r] *= a; o3[r] *= a;
    }

    // ---- O(16x64) += P(16x32) @ V(32x64)
    o0 = wmma_bf16(pu.v, vb0, o0);
    o1 = wmma_bf16(pu.v, vb1, o1);
    o2 = wmma_bf16(pu.v, vb2, o2);
    o3 = wmma_bf16(pu.v, vb3, o3);
  }

  // ---- normalize and store to [B,T,E] bf16 (heads re-interleaved)
  float inv = 1.0f / l_prev;              // valid per lane for query lrow
#pragma unroll
  for (int r = 0; r < 8; ++r) {
    int row = r + 8 * half;
    float iv = __shfl(inv, row, 32);
    int t = q0 + row;
    __bf16* orow = ob + ((size_t)(b * T_ + t)) * E_ + h * HD_ + lrow;
    orow[0]  = (__bf16)(o0[r] * iv);
    orow[16] = (__bf16)(o1[r] * iv);
    orow[32] = (__bf16)(o2[r] * iv);
    orow[48] = (__bf16)(o3[r] * iv);
  }
}

// ------------------------------------------------- 3) output projection (WMMA)
// y[m][n] = sum_e attn[m][e] * wt[n][e] + bias[n]; M=8192,N=1024,K=1024
__launch_bounds__(128)
__global__ void msa_proj_kernel(const __bf16* __restrict__ ab,
                                const __bf16* __restrict__ wt,
                                const float* __restrict__ bias,
                                float* __restrict__ out) {
  const int tid  = threadIdx.x;
  const int w    = tid >> 5;
  const int lane = tid & 31;
  const int half = lane >> 4;
  const int lrow = lane & 15;
  const int blk  = blockIdx.x;             // 512 * 16 blocks
  const int m0 = (blk & 511) * 16;
  const int n0 = (blk >> 9) * 64 + w * 16;

  const __bf16* arow = ab + (size_t)(m0 + lrow) * E_;
  v8f acc = {};
#pragma unroll 4
  for (int k0 = 0; k0 < E_; k0 += 32) {
    v16bf af = load_afrag(arow, k0, half);
    v16bf bf = load_afrag(wt + (size_t)(n0 + lrow) * E_, k0, half);
    acc = wmma_bf16(af, bf, acc);
  }
  int n = n0 + lrow;
  float bval = bias[n];
#pragma unroll
  for (int r = 0; r < 8; ++r) {
    int m = m0 + r + 8 * half;
    out[(size_t)m * E_ + n] = acc[r] + bval;
  }
}

// ---------------------------------------------------------------------- launch
extern "C" void kernel_launch(void* const* d_in, const int* in_sizes, int n_in,
                              void* d_out, int out_size, void* d_ws, size_t ws_size,
                              hipStream_t stream) {
  const float* x     = (const float*)d_in[0];
  const float* Wqkv  = (const float*)d_in[1];
  const float* bqkv  = (const float*)d_in[2];
  const float* Wproj = (const float*)d_in[3];
  const float* bproj = (const float*)d_in[4];
  float* out = (float*)d_out;

  char* ws = (char*)d_ws;
  const size_t qkv_sz = (size_t)B_ * H_ * T_ * HD_ * sizeof(__bf16);  // 16 MB
  __bf16* qb  = (__bf16*)(ws);
  __bf16* kb  = (__bf16*)(ws + qkv_sz);
  __bf16* vb  = (__bf16*)(ws + 2 * qkv_sz);
  __bf16* ob  = (__bf16*)(ws + 3 * qkv_sz);                           // 16 MB
  __bf16* wt  = (__bf16*)(ws + 4 * qkv_sz);                           //  2 MB
  __bf16* xbf = (__bf16*)(ws + 4 * qkv_sz + (size_t)E_ * E_ * sizeof(__bf16));
  __bf16* wqt = (__bf16*)(ws + 5 * qkv_sz + (size_t)E_ * E_ * sizeof(__bf16));

  msa_xbf_kernel<<<(B_ * T_ * E_) / 256, 256, 0, stream>>>(x, xbf);
  msa_wqt_kernel<<<(192 * HD_) / 256, 256, 0, stream>>>(Wqkv, wqt);
  msa_wt_kernel<<<(E_ * E_) / 256, 256, 0, stream>>>(Wproj, wt);
  msa_qkv_kernel<<<(B_ * H_) * (T_ / 64), 128, 0, stream>>>(xbf, wqt, bqkv,
                                                            qb, kb, vb);
  msa_attn_kernel<<<B_ * H_ * (T_ / 64), 128, 0, stream>>>(qb, kb, vb, ob);
  msa_proj_kernel<<<(B_ * T_ / 16) * (E_ / 64), 128, 0, stream>>>(ob, wt, bproj,
                                                                  out);
}
